// get_model_14869176778919
// MI455X (gfx1250) — matrix operand order repaired and checked by reading, not compiled, not run on hardware
//
#include <hip/hip_runtime.h>

typedef __attribute__((ext_vector_type(16))) _Float16 v16h;
typedef __attribute__((ext_vector_type(8)))  float    v8f;

#define NB   8
#define NP   1024
#define KNN  20
#define CAT  512

__device__ __forceinline__ float lrelu(float y) { return y >= 0.f ? y : 0.2f * y; }

// ---------------- f32 -> f16 weight conversion ----------------
__global__ void cvt_f16(const float* __restrict__ in, _Float16* __restrict__ out, int n) {
    int i = blockIdx.x * 256 + threadIdx.x;
    if (i < n) out[i] = (_Float16)in[i];
}

// ---------------- transpose x (B,3,N) f32 -> (B,N,4) f16 ----------------
__global__ void xpose_kernel(const float* __restrict__ x, _Float16* __restrict__ xh) {
    int t = blockIdx.x * 256 + threadIdx.x;
    if (t >= NB * NP) return;
    int b = t >> 10, n = t & (NP - 1);
    const float* xb = x + (size_t)b * 3 * NP + n;
    _Float16* o = xh + (size_t)t * 4;
    o[0] = (_Float16)xb[0];
    o[1] = (_Float16)xb[NP];
    o[2] = (_Float16)xb[2 * NP];
    o[3] = (_Float16)0.f;
}

// ---------------- per-point squared norms ----------------
template <int C, int STRIDE>
__global__ void sqnorm_kernel(const _Float16* __restrict__ xh, float* __restrict__ xx) {
    int t = blockIdx.x * 256 + threadIdx.x;
    if (t >= NB * NP) return;
    const _Float16* p = xh + (size_t)t * STRIDE;
    float s = 0.f;
#pragma unroll
    for (int c = 0; c < C; ++c) { float v = (float)p[c]; s = fmaf(v, v, s); }
    xx[t] = s;
}

// ---------------- WMMA Gram-matrix kNN + wave-level top-K ----------------
// One block per (b, 16-point row tile). 8 waves: Gram tiles via WMMA into 16x1024 LDS,
// then each wave selects top-20 for 2 rows with shuffle-argmax (no block barriers).
template <int C, int STRIDE>
__global__ void knn_wmma_kernel(const _Float16* __restrict__ xh,
                                const float* __restrict__ xx,
                                int* __restrict__ idxOut) {
    constexpr int KC = (C + 31) >> 5;
    __shared__ float dist[16][NP];          // 64 KB of the 320 KB WGP LDS
    int blk = blockIdx.x;                   // b*64 + row-tile
    int b = blk >> 6, it = blk & 63;
    int wv = threadIdx.x >> 5;
    int lane = threadIdx.x & 31;
    int m16 = lane & 15, half = lane >> 4;
    const _Float16* xb = xh + (size_t)b * NP * STRIDE;
    const float* xxb = xx + (size_t)b * NP;
    int iPoint = (it << 4) + m16;
    const _Float16* xi = xb + (size_t)iPoint * STRIDE;

    // A tile (16 center points x C), register-resident across all column tiles
    v16h a[KC];
#pragma unroll
    for (int ch = 0; ch < KC; ++ch) {
#pragma unroll
        for (int e = 0; e < 16; ++e) {
            int kk = (ch << 5) + ((e < 8) ? ((half << 3) + e) : (16 + (half << 3) + (e - 8)));
            a[ch][e] = (kk < C) ? xi[kk] : (_Float16)0.f;
        }
    }
    // norms of this lane's C/D rows (row = r + 8*half)
    float xxi[8];
#pragma unroll
    for (int r = 0; r < 8; ++r) xxi[r] = xxb[(it << 4) + r + (half << 3)];

    for (int qq = 0; qq < 8; ++qq) {
        int ct = wv + (qq << 3);            // column tile 0..63
        int mPoint = (ct << 4) + m16;
        const _Float16* xm = xb + (size_t)mPoint * STRIDE;
        v8f acc = {0.f, 0.f, 0.f, 0.f, 0.f, 0.f, 0.f, 0.f};
#pragma unroll
        for (int ch = 0; ch < KC; ++ch) {
            v16h bv;
#pragma unroll
            for (int e = 0; e < 16; ++e) {
                int k = (ch << 5) + ((e < 8) ? ((half << 3) + e) : (16 + (half << 3) + (e - 8)));
                bv[e] = (k < C) ? xm[k] : (_Float16)0.f;
            }
            acc = __builtin_amdgcn_wmma_f32_16x16x32_f16(false, a[ch], false, bv,
                                                         (short)0, acc, false, false);
        }
        float xxm = xxb[mPoint];
#pragma unroll
        for (int r = 0; r < 8; ++r) {
            int row = r + (half << 3);
            dist[row][(ct << 4) + m16] = fmaf(2.f, acc[r], -xxi[r] - xxm);
        }
    }
    __syncthreads();

    // top-K: wave wv owns rows 2wv and 2wv+1; lane owns stripe m = 32q + lane
    for (int rr = 0; rr < 2; ++rr) {
        int row = (wv << 1) + rr;
        float bv = -3e38f; int bi = NP;
        for (int q = 0; q < 32; ++q) {
            int m = (q << 5) + lane;
            float v = dist[row][m];
            if (v > bv || (v == bv && m < bi)) { bv = v; bi = m; }
        }
        int* orow = idxOut + ((size_t)b * NP + (it << 4) + row) * KNN;
        for (int sel = 0; sel < KNN; ++sel) {
            float wvv = bv; int wii = bi;
#pragma unroll
            for (int off = 16; off > 0; off >>= 1) {
                float ov = __shfl_xor(wvv, off, 32);
                int   oi = __shfl_xor(wii, off, 32);
                if (ov > wvv || (ov == wvv && oi < wii)) { wvv = ov; wii = oi; }
            }
            if (lane == 0) orow[sel] = wii;
            if ((wii & 31) == lane) {       // owner stomps + rescans its stripe only
                dist[row][wii] = -3e38f;
                bv = -3e38f; bi = NP;
                for (int q = 0; q < 32; ++q) {
                    int m = (q << 5) + lane;
                    float v = dist[row][m];
                    if (v > bv || (v == bv && m < bi)) { bv = v; bi = m; }
                }
            }
        }
    }
}

// ---------------- fused edge-conv: gather + WMMA GEMM + BN/LeakyReLU + max over k ----------------
template <int C, int STRIDE>
__global__ void edgeconv_kernel(const _Float16* __restrict__ xin,
                                const int* __restrict__ idx,
                                const _Float16* __restrict__ wh,   // (O, 2C) f16
                                const float* __restrict__ gamma, const float* __restrict__ beta,
                                const float* __restrict__ mean,  const float* __restrict__ var,
                                _Float16* __restrict__ out,        // xcat + chanOffset, stride CAT
                                int O) {
    constexpr int C2 = 2 * C;
    constexpr int KC = (C2 + 31) >> 5;
    int wave = blockIdx.x * (blockDim.x >> 5) + (threadIdx.x >> 5);
    int lane = threadIdx.x & 31;
    int tilesO = O >> 4;
    int perB = (NP >> 4) * tilesO;
    int b = wave / perB;
    if (b >= NB) return;
    int r0 = wave % perB;
    int nt = r0 / tilesO, ot = r0 % tilesO;

    int m16 = lane & 15, half = lane >> 4;
    int iPoint = (nt << 4) + m16;
    int oCh    = (ot << 4) + m16;
    const _Float16* xb = xin + (size_t)b * NP * STRIDE;
    const _Float16* xi = xb + (size_t)iPoint * STRIDE;
    const int* irow = idx + ((size_t)b * NP + iPoint) * KNN;

    float scale = gamma[oCh] * rsqrtf(var[oCh] + 1e-5f);
    float shift = beta[oCh] - mean[oCh] * scale;

    v16h bw[KC];
#pragma unroll
    for (int ch = 0; ch < KC; ++ch) {
        v16h tb;
#pragma unroll
        for (int e = 0; e < 16; ++e) {
            int k = (ch << 5) + ((e < 8) ? ((half << 3) + e) : (16 + (half << 3) + (e - 8)));
            tb[e] = (k < C2) ? wh[(size_t)oCh * C2 + k] : (_Float16)0.f;
        }
        bw[ch] = tb;
    }

    float mx[8];
#pragma unroll
    for (int r = 0; r < 8; ++r) mx[r] = -3e38f;

    for (int kn = 0; kn < KNN; ++kn) {
        int j = irow[kn];
        const _Float16* xj = xb + (size_t)j * STRIDE;
        v8f acc = {0.f, 0.f, 0.f, 0.f, 0.f, 0.f, 0.f, 0.f};
#pragma unroll
        for (int ch = 0; ch < KC; ++ch) {
            v16h a;
#pragma unroll
            for (int e = 0; e < 16; ++e) {
                int kk = (ch << 5) + ((e < 8) ? ((half << 3) + e) : (16 + (half << 3) + (e - 8)));
                _Float16 val;
                if (kk < C)       val = xj[kk] - xi[kk];
                else if (kk < C2) val = xi[kk - C];
                else              val = (_Float16)0.f;
                a[e] = val;
            }
            acc = __builtin_amdgcn_wmma_f32_16x16x32_f16(false, a, false, bw[ch],
                                                         (short)0, acc, false, false);
        }
#pragma unroll
        for (int r = 0; r < 8; ++r) {
            float y = lrelu(fmaf(acc[r], scale, shift));
            mx[r] = fmaxf(mx[r], y);
        }
    }
#pragma unroll
    for (int r = 0; r < 8; ++r) {
        int p = (nt << 4) + r + (half << 3);
        out[((size_t)b * NP + p) * CAT + oCh] = (_Float16)mx[r];
    }
}

// ---------------- final 512->1024 conv + BN/act + deterministic partial max/sum ----------------
__global__ void finalconv_kernel(const _Float16* __restrict__ xc,
                                 const _Float16* __restrict__ w5,
                                 const float* __restrict__ gamma, const float* __restrict__ beta,
                                 const float* __restrict__ mean,  const float* __restrict__ var,
                                 float* __restrict__ pmax, float* __restrict__ psum) {
    int wave = blockIdx.x * (blockDim.x >> 5) + (threadIdx.x >> 5);
    int lane = threadIdx.x & 31;
    const int tilesO = 64;
    const int perB = 64 * 64;
    int b = wave / perB;
    if (b >= NB) return;
    int r0 = wave % perB;
    int nt = r0 / tilesO, ot = r0 % tilesO;
    int m16 = lane & 15, half = lane >> 4;
    int iPoint = (nt << 4) + m16;
    int oCh    = (ot << 4) + m16;
    const _Float16* arow = xc + ((size_t)b * NP + iPoint) * CAT;
    float scale = gamma[oCh] * rsqrtf(var[oCh] + 1e-5f);
    float shift = beta[oCh] - mean[oCh] * scale;

    v8f acc = {0.f, 0.f, 0.f, 0.f, 0.f, 0.f, 0.f, 0.f};
    for (int ch = 0; ch < 16; ++ch) {
        v16h a, bv;
#pragma unroll
        for (int e = 0; e < 16; ++e) {
            int kk = (ch << 5) + ((e < 8) ? ((half << 3) + e) : (16 + (half << 3) + (e - 8)));
            a[e] = arow[kk];
        }
#pragma unroll
        for (int e = 0; e < 16; ++e) {
            int k = (ch << 5) + ((e < 8) ? ((half << 3) + e) : (16 + (half << 3) + (e - 8)));
            bv[e] = w5[(size_t)oCh * CAT + k];
        }
        acc = __builtin_amdgcn_wmma_f32_16x16x32_f16(false, a, false, bv,
                                                     (short)0, acc, false, false);
    }
    float lmax = -3e38f, lsum = 0.f;
#pragma unroll
    for (int r = 0; r < 8; ++r) {
        float y = lrelu(fmaf(acc[r], scale, shift));
        lmax = fmaxf(lmax, y);
        lsum += y;
    }
    float om = __shfl_xor(lmax, 16, 32);
    float os = __shfl_xor(lsum, 16, 32);
    lmax = fmaxf(lmax, om);
    lsum += os;
    if (half == 0) {
        size_t o = ((size_t)b * 1024 + oCh) * 64 + nt;
        pmax[o] = lmax;
        psum[o] = lsum;
    }
}

// ---------------- fold partials -> output (B, 2048) = [max(1024) | mean(1024)] ----------------
__global__ void reduce_kernel(const float* __restrict__ pmax, const float* __restrict__ psum,
                              float* __restrict__ out) {
    int t = blockIdx.x * 256 + threadIdx.x;
    if (t >= NB * 1024) return;
    int b = t >> 10, o = t & 1023;
    const float* pm = pmax + (size_t)t * 64;
    const float* ps = psum + (size_t)t * 64;
    float mx = -3e38f, sm = 0.f;
    for (int q = 0; q < 64; ++q) { mx = fmaxf(mx, pm[q]); sm += ps[q]; }
    out[(size_t)b * 2048 + o]        = mx;
    out[(size_t)b * 2048 + 1024 + o] = sm * (1.0f / 1024.0f);
}

extern "C" void kernel_launch(void* const* d_in, const int* in_sizes, int n_in,
                              void* d_out, int out_size, void* d_ws, size_t ws_size,
                              hipStream_t stream) {
    (void)in_sizes; (void)n_in; (void)out_size; (void)ws_size;
    const float* x = (const float*)d_in[0];
    const float *w[5], *g[5], *bb[5], *mn[5], *vr[5];
    for (int i = 0; i < 5; ++i) {
        w[i]  = (const float*)d_in[1 + 5 * i + 0];
        g[i]  = (const float*)d_in[1 + 5 * i + 1];
        bb[i] = (const float*)d_in[1 + 5 * i + 2];
        mn[i] = (const float*)d_in[1 + 5 * i + 3];
        vr[i] = (const float*)d_in[1 + 5 * i + 4];
    }
    char* p = (char*)d_ws;
    auto carve = [&](size_t bytes) -> char* {
        char* r = p; p += (bytes + 255) & ~(size_t)255; return r;
    };
    _Float16* xh0  = (_Float16*)carve((size_t)NB * NP * 4 * 2);
    _Float16* xcat = (_Float16*)carve((size_t)NB * NP * CAT * 2);
    int*      idx  = (int*)carve((size_t)NB * NP * KNN * 4);
    float*    xxb  = (float*)carve((size_t)NB * NP * 4);
    const int wsz[5] = {64 * 6, 64 * 128, 128 * 128, 256 * 256, 1024 * 512};
    _Float16* wh[5];
    for (int i = 0; i < 5; ++i) wh[i] = (_Float16*)carve((size_t)wsz[i] * 2);
    float* pmax = (float*)carve((size_t)NB * 1024 * 64 * 4);
    float* psum = (float*)carve((size_t)NB * 1024 * 64 * 4);

    for (int i = 0; i < 5; ++i)
        cvt_f16<<<(wsz[i] + 255) / 256, 256, 0, stream>>>(w[i], wh[i], wsz[i]);
    xpose_kernel<<<(NB * NP + 255) / 256, 256, 0, stream>>>(x, xh0);

    // ---- layer 1: C=3 (stride 4), O=64 -> xcat[:, :, 0:64]
    sqnorm_kernel<3, 4><<<32, 256, 0, stream>>>(xh0, xxb);
    knn_wmma_kernel<3, 4><<<NB * 64, 256, 0, stream>>>(xh0, xxb, idx);
    edgeconv_kernel<3, 4><<<(NB * 64 * 4) / 8, 256, 0, stream>>>(
        xh0, idx, wh[0], g[0], bb[0], mn[0], vr[0], xcat + 0, 64);
    // ---- layer 2: C=64, O=64 -> xcat[:, :, 64:128]
    sqnorm_kernel<64, CAT><<<32, 256, 0, stream>>>(xcat + 0, xxb);
    knn_wmma_kernel<64, CAT><<<NB * 64, 256, 0, stream>>>(xcat + 0, xxb, idx);
    edgeconv_kernel<64, CAT><<<(NB * 64 * 4) / 8, 256, 0, stream>>>(
        xcat + 0, idx, wh[1], g[1], bb[1], mn[1], vr[1], xcat + 64, 64);
    // ---- layer 3: C=64, O=128 -> xcat[:, :, 128:256]
    sqnorm_kernel<64, CAT><<<32, 256, 0, stream>>>(xcat + 64, xxb);
    knn_wmma_kernel<64, CAT><<<NB * 64, 256, 0, stream>>>(xcat + 64, xxb, idx);
    edgeconv_kernel<64, CAT><<<(NB * 64 * 8) / 8, 256, 0, stream>>>(
        xcat + 64, idx, wh[2], g[2], bb[2], mn[2], vr[2], xcat + 128, 128);
    // ---- layer 4: C=128, O=256 -> xcat[:, :, 256:512]
    sqnorm_kernel<128, CAT><<<32, 256, 0, stream>>>(xcat + 128, xxb);
    knn_wmma_kernel<128, CAT><<<NB * 64, 256, 0, stream>>>(xcat + 128, xxb, idx);
    edgeconv_kernel<128, CAT><<<(NB * 64 * 16) / 8, 256, 0, stream>>>(
        xcat + 128, idx, wh[3], g[3], bb[3], mn[3], vr[3], xcat + 256, 256);

    // ---- final conv 512 -> 1024 + pooling partials
    finalconv_kernel<<<(NB * 64 * 64) / 8, 256, 0, stream>>>(
        xcat, wh[4], g[4], bb[4], mn[4], vr[4], pmax, psum);
    reduce_kernel<<<(NB * 1024 + 255) / 256, 256, 0, stream>>>(pmax, psum, (float*)d_out);
}
